// DGGGL_80977313399323
// MI455X (gfx1250) — matrix operand order repaired and checked by reading, not compiled, not run on hardware
//
#include <hip/hip_runtime.h>
#include <math.h>

// ---- problem constants ----
#define B_   8
#define N_   2048
#define NSUP 2
#define NPAD 80            // 66 features padded to 80 (5 x 16 col tiles, 160B rows)

typedef __attribute__((ext_vector_type(16))) __bf16 v16bf;
typedef __attribute__((ext_vector_type(8)))  float  v8f;

static __device__ __forceinline__ float sigf(float x) { return 1.0f / (1.0f + __expf(-x)); }

// =====================================================================
// Kernel 1: mr = sigmoid(gate_in @ mlp_w + b); state = mr*s1+(1-mr)*s2;
//           Xpad(bf16, B,N,80) = [xt | s1 | s2 | 0-pad]
// =====================================================================
__global__ __launch_bounds__(256) void prep_kernel(
    const float* __restrict__ xt, const float* __restrict__ s1,
    const float* __restrict__ s2, const float* __restrict__ ge,
    const float* __restrict__ mlp_w, const float* __restrict__ mlp_b,
    float* __restrict__ state, unsigned short* __restrict__ Xpad_u)
{
  __bf16* Xpad = (__bf16*)Xpad_u;
  const int row = blockIdx.x * 8 + threadIdx.y;     // 0 .. B*N-1
  const int o   = threadIdx.x;                       // 0 .. 31
  float acc = mlp_b[o];
  for (int c = 0; c < 74; ++c) {
    float v;
    if      (c < 2)  v = xt[row * 2  + c];
    else if (c < 34) v = s1[row * 32 + (c - 2)];
    else if (c < 66) v = s2[row * 32 + (c - 34)];
    else             v = ge[row * 8  + (c - 66)];
    acc += v * mlp_w[c * 32 + o];
  }
  const float mr = sigf(acc);
  state[row * 32 + o] = mr * s1[row * 32 + o] + (1.0f - mr) * s2[row * 32 + o];

  const size_t rb = (size_t)row * NPAD;
  for (int c = o; c < NPAD; c += 32) {
    float v = 0.0f;
    if      (c < 2)  v = xt[row * 2  + c];
    else if (c < 34) v = s1[row * 32 + (c - 2)];
    else if (c < 66) v = s2[row * 32 + (c - 34)];
    Xpad[rb + c] = (__bf16)v;
  }
}

// =====================================================================
// Kernel 2: heavy GEMM  acc(2048x80) = A(2048x2048) @ Tin(2048x80,bf16)
//  AMODE 0: A from f32 (convert in-register)
//  AMODE 1: A from f32, also store bf16 image of A (convert-once pass)
//  AMODE 2: A from the bf16 image (half the bytes, no cvt chain)
//  CHEB2 0: Tout = acc        (t1 = A @ X)
//  CHEB2 1: Tout = 2*acc - Tprev  (t2 = 2 A t1 - t0)
// grid (16, B, NSUP), 256 thr = 8 waves; wave = 16 rows x 80 cols.
// Fragments double-buffered across K-steps; 5 WMMA per 32-wide K step.
// =====================================================================
template<int AMODE, int CHEB2>
__global__ __launch_bounds__(256) void cheb_gemm_kernel(
    const float* __restrict__ Af32,
    const unsigned short* __restrict__ AbfIn_u,
    unsigned short* __restrict__ AbfOut_u,
    const unsigned short* __restrict__ Tin_u,
    const unsigned short* __restrict__ Tprev_u,
    unsigned short* __restrict__ Tout_u,
    int tinHasS)
{
  const int b    = blockIdx.y;
  const int s    = blockIdx.z;
  const int wave = threadIdx.x >> 5;
  const int lane = threadIdx.x & 31;
  const int half = lane >> 4;
  const int m    = lane & 15;
  const int rowBase = blockIdx.x * 128 + wave * 16;

  // 16-bit A layout (16x32): lanes 0-15: K={0..7,16..23}; lanes 16-31: K={8..15,24..31}
  const size_t aoff   = ((size_t)(s * B_ + b)) * N_ * N_ + (size_t)(rowBase + m) * N_ + 8 * half;
  const float*  Arow    = Af32 + aoff;
  const __bf16* AbfRow  = (const __bf16*)AbfIn_u + aoff;
  __bf16*       AbfSt   = (__bf16*)AbfOut_u + aoff;

  const __bf16* TinB   = (const __bf16*)Tin_u  + ((size_t)(tinHasS ? (s * B_ + b) : b)) * N_ * NPAD;
  const __bf16* TprevB = (const __bf16*)Tprev_u + ((size_t)b) * N_ * NPAD;
  __bf16*       ToutB  = (__bf16*)Tout_u + ((size_t)(s * B_ + b)) * N_ * NPAD;

  v8f acc[5];
  const v8f zero = {0.f, 0.f, 0.f, 0.f, 0.f, 0.f, 0.f, 0.f};
#pragma unroll
  for (int t = 0; t < 5; ++t) acc[t] = zero;

  struct AF { float4 f[4]; uint4 q[2]; };
  union  BU { uint4 q[2]; v16bf v; };

  auto loadA = [&](int k0, AF& a) {
    if (AMODE == 2) {
      a.q[0] = *(const uint4*)(AbfRow + k0);
      a.q[1] = *(const uint4*)(AbfRow + k0 + 16);
      if (k0 + 512 < N_) __builtin_prefetch(AbfRow + k0 + 512, 0, 1);
    } else {
      a.f[0] = *(const float4*)(Arow + k0 + 0);
      a.f[1] = *(const float4*)(Arow + k0 + 4);
      a.f[2] = *(const float4*)(Arow + k0 + 16);
      a.f[3] = *(const float4*)(Arow + k0 + 20);
      if (k0 + 512 < N_) __builtin_prefetch(Arow + k0 + 512, 0, 1);
    }
  };
  auto loadB = [&](int k0, BU* bu) {
    const __bf16* Brow = TinB + (size_t)k0 * NPAD + (lane & 15);
#pragma unroll
    for (int nt = 0; nt < 5; ++nt) {
#pragma unroll
      for (int e = 0; e < 16; ++e) {
        const int kk = (e < 8) ? (8 * half + e) : (16 + 8 * half + (e - 8));
        bu[nt].v[e] = Brow[(size_t)kk * NPAD + nt * 16];
      }
    }
  };
  auto compute = [&](int k0, const AF& a, const BU* bu) {
    v16bf af;
    if (AMODE == 2) {
      union { uint4 q[2]; v16bf v; } au;
      au.q[0] = a.q[0]; au.q[1] = a.q[1];
      af = au.v;
    } else {
      af[0]  = (__bf16)a.f[0].x; af[1]  = (__bf16)a.f[0].y; af[2]  = (__bf16)a.f[0].z; af[3]  = (__bf16)a.f[0].w;
      af[4]  = (__bf16)a.f[1].x; af[5]  = (__bf16)a.f[1].y; af[6]  = (__bf16)a.f[1].z; af[7]  = (__bf16)a.f[1].w;
      af[8]  = (__bf16)a.f[2].x; af[9]  = (__bf16)a.f[2].y; af[10] = (__bf16)a.f[2].z; af[11] = (__bf16)a.f[2].w;
      af[12] = (__bf16)a.f[3].x; af[13] = (__bf16)a.f[3].y; af[14] = (__bf16)a.f[3].z; af[15] = (__bf16)a.f[3].w;
      if (AMODE == 1) {           // persist bf16 image of A for later passes
        union { v16bf v; uint4 q[2]; } au;
        au.v = af;
        *(uint4*)(AbfSt + k0)      = au.q[0];
        *(uint4*)(AbfSt + k0 + 16) = au.q[1];
      }
    }
#pragma unroll
    for (int nt = 0; nt < 5; ++nt)
      acc[nt] = __builtin_amdgcn_wmma_f32_16x16x32_bf16(
          false, af, false, bu[nt].v, (short)0, acc[nt], false, false);
  };

  // software pipeline: double-buffered A and B fragments
  AF a0, a1;
  BU b0[5], b1[5];
  loadA(0, a0); loadB(0, b0);
  for (int k0 = 0; k0 < N_ - 64; k0 += 64) {
    loadA(k0 + 32, a1); loadB(k0 + 32, b1);
    compute(k0, a0, b0);
    loadA(k0 + 64, a0); loadB(k0 + 64, b0);
    compute(k0 + 32, a1, b1);
  }
  {
    const int k0 = N_ - 64;
    loadA(k0 + 32, a1); loadB(k0 + 32, b1);
    compute(k0, a0, b0);
    compute(k0 + 32, a1, b1);
  }

  // C/D layout: lane = N col (0-15 twice), VGPR r -> row r (+8 for lanes 16-31)
  const int col0 = lane & 15;
  const int rOff = 8 * half;
#pragma unroll
  for (int nt = 0; nt < 5; ++nt) {
#pragma unroll
    for (int r = 0; r < 8; ++r) {
      const size_t idx = (size_t)(rowBase + rOff + r) * NPAD + nt * 16 + col0;
      float v = acc[nt][r];
      if (CHEB2) v = 2.0f * v - (float)TprevB[idx];
      ToutB[idx] = (__bf16)v;
    }
  }
}

// =====================================================================
// Kernel 3: basis matmul x_g @ W + b, activation, gating.
//  mode 0 (gate): 96 outs -> z1,z2,r ; store r ; build candidate Cpad (bf16)
//  mode 1 (upd) : 32 outs -> hc=tanh ; h = r*state+(1-r)*hc ; transnext = h@hop_w+b
// =====================================================================
__global__ __launch_bounds__(256) void combine_kernel(
    const unsigned short* __restrict__ T0_u,
    const unsigned short* __restrict__ t1_u,
    const unsigned short* __restrict__ t2_u,
    const float* __restrict__ W, const float* __restrict__ bias,
    const float* __restrict__ xt, const float* __restrict__ s1,
    const float* __restrict__ s2,
    const float* __restrict__ statebuf, float* __restrict__ rbuf,
    unsigned short* __restrict__ Cpad_u,
    const float* __restrict__ hop_w, const float* __restrict__ hop_b,
    float* __restrict__ out, int mode)
{
  __shared__ float hrow[8][32];
  const int row = blockIdx.x * 8 + threadIdx.y;
  const int o   = threadIdx.x;
  const size_t rb   = (size_t)row * NPAD;
  const size_t SSTR = (size_t)B_ * N_ * NPAD;     // support stride in t1/t2

  const __bf16* T0 = (const __bf16*)T0_u;
  const __bf16* t1 = (const __bf16*)t1_u;
  const __bf16* t2 = (const __bf16*)t2_u;
  const __bf16* tp[6] = { T0 + rb, t1 + rb, t2 + rb,
                          T0 + rb, t1 + SSTR + rb, t2 + SSTR + rb };

  if (mode == 0) {
    float a0 = bias[o], a1 = bias[o + 32], a2 = bias[o + 64];
    for (int ti = 0; ti < 6; ++ti) {
      const __bf16* T  = tp[ti];
      const float*  Wr = W + (size_t)ti * 66 * 96;
      for (int c = 0; c < 66; ++c) {
        const float tv = (float)T[c];
        const float* w = Wr + c * 96;
        a0 += tv * w[o]; a1 += tv * w[o + 32]; a2 += tv * w[o + 64];
      }
    }
    const float z1 = sigf(a0), z2 = sigf(a1), r = sigf(a2);
    rbuf[row * 32 + o] = r;
    __bf16* Cpad = (__bf16*)Cpad_u;
    Cpad[rb + 2 + o]  = (__bf16)(z1 * s1[row * 32 + o]);
    Cpad[rb + 34 + o] = (__bf16)(z2 * s2[row * 32 + o]);
    if (o < 2)  Cpad[rb + o]      = (__bf16)xt[row * 2 + o];
    if (o < 14) Cpad[rb + 66 + o] = (__bf16)0.0f;
  } else {
    float a = bias[o];
    for (int ti = 0; ti < 6; ++ti) {
      const __bf16* T  = tp[ti];
      const float*  Wr = W + (size_t)ti * 66 * 32;
      for (int c = 0; c < 66; ++c)
        a += (float)T[c] * Wr[c * 32 + o];
    }
    const float hc = tanhf(a);
    const float r  = rbuf[row * 32 + o];
    const float h  = r * statebuf[row * 32 + o] + (1.0f - r) * hc;
    out[(size_t)row * 32 + o] = h;
    hrow[threadIdx.y][o] = h;
    __syncthreads();
    float tn = hop_b[o];
    for (int c = 0; c < 32; ++c)
      tn += hrow[threadIdx.y][c] * hop_w[c * 32 + o];
    out[(size_t)B_ * N_ * 32 + (size_t)row * 32 + o] = tn;
  }
}

// =====================================================================
extern "C" void kernel_launch(void* const* d_in, const int* in_sizes, int n_in,
                              void* d_out, int out_size, void* d_ws, size_t ws_size,
                              hipStream_t stream)
{
  (void)in_sizes; (void)n_in; (void)out_size;
  const float* xt     = (const float*)d_in[0];
  const float* s1     = (const float*)d_in[1];
  const float* s2     = (const float*)d_in[2];
  const float* ge     = (const float*)d_in[3];
  const float* sup    = (const float*)d_in[4];
  const float* mlp_w  = (const float*)d_in[5];
  const float* mlp_b  = (const float*)d_in[6];
  const float* gate_w = (const float*)d_in[7];
  const float* gate_b = (const float*)d_in[8];
  const float* upd_w  = (const float*)d_in[9];
  const float* upd_b  = (const float*)d_in[10];
  const float* hop_w  = (const float*)d_in[11];
  const float* hop_b  = (const float*)d_in[12];
  float* out = (float*)d_out;

  // workspace carve: ~20 MB core + optional 134 MB bf16 image of supports
  char* ws = (char*)d_ws;
  size_t off = 0;
  auto carve = [&](size_t bytes) -> void* {
    void* p = ws + off;
    off += (bytes + 255) & ~(size_t)255;
    return p;
  };
  unsigned short* Xpad = (unsigned short*)carve((size_t)B_ * N_ * NPAD * 2);
  unsigned short* Cpad = (unsigned short*)carve((size_t)B_ * N_ * NPAD * 2);
  unsigned short* t1   = (unsigned short*)carve((size_t)NSUP * B_ * N_ * NPAD * 2);
  unsigned short* t2   = (unsigned short*)carve((size_t)NSUP * B_ * N_ * NPAD * 2);
  float* state = (float*)carve((size_t)B_ * N_ * 32 * 4);
  float* rbuf  = (float*)carve((size_t)B_ * N_ * 32 * 4);
  const size_t abfBytes = (size_t)NSUP * B_ * N_ * N_ * 2;   // 134 MB
  const size_t needBig  = off + abfBytes;
  unsigned short* Abf = (unsigned short*)carve(abfBytes);
  const bool big = (ws_size >= needBig);

  const dim3 blkRows(32, 8);
  const int  nRowBlocks = (B_ * N_) / 8;     // 2048
  const dim3 gGemm(N_ / 128, B_, NSUP);      // (16, 8, 2)

  // 1) mr / state / Xpad
  prep_kernel<<<nRowBlocks, blkRows, 0, stream>>>(xt, s1, s2, ge, mlp_w, mlp_b, state, Xpad);

  if (big) {
    // pass 1 converts A->bf16 once; passes 2-4 read the (mostly L2-resident) bf16 image
    cheb_gemm_kernel<1,0><<<gGemm, 256, 0, stream>>>(sup, Abf, Abf, Xpad, Xpad, t1, 0);
    cheb_gemm_kernel<2,1><<<gGemm, 256, 0, stream>>>(sup, Abf, Abf, t1,   Xpad, t2, 1);
    combine_kernel<<<nRowBlocks, blkRows, 0, stream>>>(Xpad, t1, t2, gate_w, gate_b,
        xt, s1, s2, state, rbuf, Cpad, hop_w, hop_b, out, 0);
    cheb_gemm_kernel<2,0><<<gGemm, 256, 0, stream>>>(sup, Abf, Abf, Cpad, Cpad, t1, 0);
    cheb_gemm_kernel<2,1><<<gGemm, 256, 0, stream>>>(sup, Abf, Abf, t1,   Cpad, t2, 1);
    combine_kernel<<<nRowBlocks, blkRows, 0, stream>>>(Cpad, t1, t2, upd_w, upd_b,
        xt, s1, s2, state, rbuf, Cpad, hop_w, hop_b, out, 1);
  } else {
    // fallback: stream f32 supports every pass
    cheb_gemm_kernel<0,0><<<gGemm, 256, 0, stream>>>(sup, Abf, Abf, Xpad, Xpad, t1, 0);
    cheb_gemm_kernel<0,1><<<gGemm, 256, 0, stream>>>(sup, Abf, Abf, t1,   Xpad, t2, 1);
    combine_kernel<<<nRowBlocks, blkRows, 0, stream>>>(Xpad, t1, t2, gate_w, gate_b,
        xt, s1, s2, state, rbuf, Cpad, hop_w, hop_b, out, 0);
    cheb_gemm_kernel<0,0><<<gGemm, 256, 0, stream>>>(sup, Abf, Abf, Cpad, Cpad, t1, 0);
    cheb_gemm_kernel<0,1><<<gGemm, 256, 0, stream>>>(sup, Abf, Abf, t1,   Cpad, t2, 1);
    combine_kernel<<<nRowBlocks, blkRows, 0, stream>>>(Cpad, t1, t2, upd_w, upd_b,
        xt, s1, s2, state, rbuf, Cpad, hop_w, hop_b, out, 1);
  }
}
